// SimpleVQAutoEncoder_70652212019550
// MI455X (gfx1250) — matrix operand order repaired and checked by reading, not compiled, not run on hardware
//
#include <hip/hip_runtime.h>
#include <hip/hip_bf16.h>

typedef __attribute__((ext_vector_type(16))) __bf16 v16bf;
typedef __attribute__((ext_vector_type(8)))  __bf16 v8bf;
typedef __attribute__((ext_vector_type(8)))  float  v8f;
typedef int v4i_gcc __attribute__((vector_size(4 * sizeof(int))));

#define TILE_M 128
#define TILE_N 128
#define TILE_K 32
#define LDA_S  40    // 32 + 8 pad  (80B row stride, 16B multiple -> legal B128 LDS writes)
#define LDB_S  136   // 128 + 8 pad (272B row stride, 16B multiple)

// ---------------------------------------------------------------------------
// CDNA5 async global->LDS copy (ASYNCcnt path), 16B per lane, with fallback.
// Builtin prototype (from hipcc diagnostic): (v4i AS1*, v4i AS3*, imm, imm).
// ---------------------------------------------------------------------------
__device__ __forceinline__ void async_copy_b128(void* lds_dst, const void* gsrc) {
#if __has_builtin(__builtin_amdgcn_global_load_async_to_lds_b128)
    __builtin_amdgcn_global_load_async_to_lds_b128(
        (__attribute__((address_space(1))) v4i_gcc*)(uintptr_t)gsrc,
        (__attribute__((address_space(3))) v4i_gcc*)(uint32_t)(uintptr_t)lds_dst,
        0, 0);
#else
    *(v8bf*)lds_dst = *(const v8bf*)gsrc;
#endif
}

__device__ __forceinline__ void wait_async_all() {
#if __has_builtin(__builtin_amdgcn_s_wait_asynccnt)
    __builtin_amdgcn_s_wait_asynccnt(0);
#else
    asm volatile("s_wait_asynccnt 0" ::: "memory");
#endif
}

__device__ __forceinline__ float gelu_exact(float x) {
    return 0.5f * x * (1.0f + erff(x * 0.70710678118654752f));
}

// ---------------------------------------------------------------------------
// f32 -> bf16 bulk convert (one-time per tensor; n must be a multiple of 8)
// ---------------------------------------------------------------------------
__global__ __launch_bounds__(256)
void cvt_bf16(const float* __restrict__ src, __bf16* __restrict__ dst, int nvec8) {
    int i = blockIdx.x * 256 + threadIdx.x;
    if (i >= nvec8) return;
    const float4* s = (const float4*)src + (size_t)i * 2;
    float4 a = s[0], b = s[1];
    v8bf o = {(__bf16)a.x, (__bf16)a.y, (__bf16)a.z, (__bf16)a.w,
              (__bf16)b.x, (__bf16)b.y, (__bf16)b.z, (__bf16)b.w};
    *((v8bf*)dst + i) = o;
}

// ---------------------------------------------------------------------------
// C[M,N] = act(A[M,K] @ W[K,N] + bias[N]); A,W bf16, accum f32, OutT output.
// 128x128 block tile, 8 waves (2x4), wave tile 64x32, async double-buffered LDS.
// ---------------------------------------------------------------------------
template <bool GELU_ACT, typename OutT>
__global__ __launch_bounds__(256)
void gemm_bias_act(const __bf16* __restrict__ A, const __bf16* __restrict__ W,
                   const float* __restrict__ bias, OutT* __restrict__ C,
                   int M, int N, int K) {
    __shared__ __attribute__((aligned(16))) __bf16 As[2][TILE_M * LDA_S];
    __shared__ __attribute__((aligned(16))) __bf16 Bs[2][TILE_K * LDB_S];

    const int tid  = threadIdx.x;
    const int lane = tid & 31;
    const int wid  = tid >> 5;
    const int wr   = wid >> 2;   // 0..1
    const int wc   = wid & 3;    // 0..3
    const int r    = lane & 15;
    const int g    = lane >> 4;
    const int bm0  = blockIdx.y * TILE_M;
    const int bn0  = blockIdx.x * TILE_N;

    auto stage = [&](int k0, int buf) {
        // A tile: 128x32 bf16 = 512 x 16B chunks
#pragma unroll
        for (int c = tid; c < 512; c += 256) {
            int row = c >> 2, ko = (c & 3) * 8;
            async_copy_b128(&As[buf][row * LDA_S + ko],
                            A + (size_t)(bm0 + row) * K + k0 + ko);
        }
        // B tile: 32x128 bf16 = 512 x 16B chunks
#pragma unroll
        for (int c = tid; c < 512; c += 256) {
            int kk = c >> 4, no = (c & 15) * 8;
            async_copy_b128(&Bs[buf][kk * LDB_S + no],
                            W + (size_t)(k0 + kk) * N + bn0 + no);
        }
    };

    v8f acc[4][2] = {};
    const int nk = K / TILE_K;
    stage(0, 0);

    for (int t = 0; t < nk; ++t) {
        wait_async_all();
        __syncthreads();
        if (t + 1 < nk) stage((t + 1) * TILE_K, (t + 1) & 1);

        const __bf16* as = As[t & 1];
        const __bf16* bs = Bs[t & 1];

        v16bf bfrag[2];
#pragma unroll
        for (int nf = 0; nf < 2; ++nf) {
            int col = wc * 32 + nf * 16 + r;
#pragma unroll
            for (int v = 0; v < 8; ++v)
#pragma unroll
                for (int e = 0; e < 2; ++e) {
                    int kk = ((v < 4) ? v * 2 : 16 + (v - 4) * 2) + g * 8 + e;
                    bfrag[nf][v * 2 + e] = bs[kk * LDB_S + col];
                }
        }
#pragma unroll
        for (int mf = 0; mf < 4; ++mf) {
            v16bf afrag;
            int row = wr * 64 + mf * 16 + r;
#pragma unroll
            for (int v = 0; v < 8; ++v)
#pragma unroll
                for (int e = 0; e < 2; ++e) {
                    int kk = ((v < 4) ? v * 2 : 16 + (v - 4) * 2) + g * 8 + e;
                    afrag[v * 2 + e] = as[row * LDA_S + kk];
                }
#pragma unroll
            for (int nf = 0; nf < 2; ++nf)
                acc[mf][nf] = __builtin_amdgcn_wmma_f32_16x16x32_bf16(
                    false, afrag, false, bfrag[nf], (short)0, acc[mf][nf], false, false);
        }
    }

#pragma unroll
    for (int mf = 0; mf < 4; ++mf)
#pragma unroll
        for (int nf = 0; nf < 2; ++nf) {
            int col = bn0 + wc * 32 + nf * 16 + r;
            float bv = bias[col];
#pragma unroll
            for (int v = 0; v < 8; ++v) {
                int row = bm0 + wr * 64 + mf * 16 + v + 8 * g;
                float x = acc[mf][nf][v] + bv;
                if (GELU_ACT) x = gelu_exact(x);
                C[(size_t)row * N + col] = (OutT)x;
            }
        }
}

// ---------------------------------------------------------------------------
// Codebook row norms (f32, exact)
// ---------------------------------------------------------------------------
__global__ __launch_bounds__(256)
void cnorm_kernel(const float* __restrict__ codebook, float* __restrict__ cnorm) {
    int n = blockIdx.x * 256 + threadIdx.x;
    const float4* row = (const float4*)(codebook + (size_t)n * 128);
    float s = 0.f;
#pragma unroll 8
    for (int j = 0; j < 32; ++j) {
        float4 v = row[j];
        s += v.x * v.x + v.y * v.y + v.z * v.z + v.w * v.w;
    }
    cnorm[n] = s;
}

__global__ void zero1(float* p) { *p = 0.f; }

// ---------------------------------------------------------------------------
// Fused VQ argmin: per wave 16 rows of Zb[65536,128] vs all 1024 codes (WMMA).
// dist(n) = ||c_n||^2 - 2<z,c_n>. Codebook tiles async double-buffered in LDS.
// ---------------------------------------------------------------------------
__global__ __launch_bounds__(256)
void vq_argmin(const __bf16* __restrict__ Zb, const __bf16* __restrict__ cb,
               const float* __restrict__ cnorm, int* __restrict__ indices) {
    __shared__ __attribute__((aligned(16))) __bf16 Cs[2][16 * LDB_S];

    const int tid  = threadIdx.x;
    const int lane = tid & 31;
    const int wid  = tid >> 5;
    const int r    = lane & 15;
    const int g    = lane >> 4;
    const int rows0 = blockIdx.x * 128 + wid * 16;

    // A fragments for full K=128 (4 chunks of 32) held in registers
    v16bf afr[4];
#pragma unroll
    for (int kc = 0; kc < 4; ++kc)
#pragma unroll
        for (int v = 0; v < 8; ++v)
#pragma unroll
            for (int e = 0; e < 2; ++e) {
                int kk = kc * 32 + ((v < 4) ? v * 2 : 16 + (v - 4) * 2) + g * 8 + e;
                afr[kc][v * 2 + e] = Zb[(size_t)(rows0 + r) * 128 + kk];
            }

    auto stage = [&](int n0, int buf) {
        // 16x128 bf16 = 256 x 16B chunks -> exactly one per thread
        int nr = tid >> 4, doff = (tid & 15) * 8;
        async_copy_b128(&Cs[buf][nr * LDB_S + doff],
                        cb + (size_t)(n0 + nr) * 128 + doff);
    };

    float bestd[8];
    int   bestn[8];
#pragma unroll
    for (int v = 0; v < 8; ++v) { bestd[v] = 3.4e38f; bestn[v] = 0; }

    stage(0, 0);
    for (int t = 0; t < 64; ++t) {
        wait_async_all();
        __syncthreads();
        if (t + 1 < 64) stage((t + 1) * 16, (t + 1) & 1);

        const __bf16* cs = Cs[t & 1];
        v8f acc = {};
#pragma unroll
        for (int kc = 0; kc < 4; ++kc) {
            v16bf bfr;
#pragma unroll
            for (int v = 0; v < 8; ++v)
#pragma unroll
                for (int e = 0; e < 2; ++e) {
                    int kk = kc * 32 + ((v < 4) ? v * 2 : 16 + (v - 4) * 2) + g * 8 + e;
                    bfr[v * 2 + e] = cs[r * LDB_S + kk];
                }
            acc = __builtin_amdgcn_wmma_f32_16x16x32_bf16(
                false, afr[kc], false, bfr, (short)0, acc, false, false);
        }
        int   n0 = t * 16;
        float cn = cnorm[n0 + r];
        int   n  = n0 + r;
#pragma unroll
        for (int v = 0; v < 8; ++v) {
            float d = cn - 2.0f * acc[v];
            if (d < bestd[v]) { bestd[v] = d; bestn[v] = n; }
        }
    }

    // Argmin across the 16 lanes sharing a row group (first-min tie break)
#pragma unroll
    for (int off = 1; off < 16; off <<= 1)
#pragma unroll
        for (int v = 0; v < 8; ++v) {
            float od = __shfl_xor(bestd[v], off, 32);
            int   on = __shfl_xor(bestn[v], off, 32);
            if (od < bestd[v] || (od == bestd[v] && on < bestn[v])) {
                bestd[v] = od; bestn[v] = on;
            }
        }
    if (r == 0)
#pragma unroll
        for (int v = 0; v < 8; ++v)
            indices[rows0 + v + 8 * g] = bestn[v];
}

// ---------------------------------------------------------------------------
// qb = bf16(codebook[idx]); commit loss partials (f32 quantized vs bf16 z)
// ---------------------------------------------------------------------------
__global__ __launch_bounds__(256)
void quant_commit(const __bf16* __restrict__ Zb, const float* __restrict__ codebook,
                  const int* __restrict__ indices, __bf16* __restrict__ Qb,
                  float* __restrict__ acc) {
    __shared__ float red[256];
    size_t i = (size_t)blockIdx.x * 256 + threadIdx.x;
    int row = (int)(i >> 7);
    int d   = (int)(i & 127);
    int idx = indices[row];
    float qv = codebook[(size_t)idx * 128 + d];
    Qb[i] = (__bf16)qv;
    float diff = qv - (float)Zb[i];
    red[threadIdx.x] = diff * diff;
    __syncthreads();
    for (int s = 128; s > 0; s >>= 1) {
        if ((int)threadIdx.x < s) red[threadIdx.x] += red[threadIdx.x + s];
        __syncthreads();
    }
    if (threadIdx.x == 0) atomicAdd(acc, red[0]);
}

__global__ __launch_bounds__(256)
void finalize(const int* __restrict__ indices, const float* __restrict__ acc,
              float* __restrict__ out_idx, float* __restrict__ out_loss) {
    int i = blockIdx.x * 256 + threadIdx.x;
    if (i < 65536) out_idx[i] = (float)indices[i];
    if (i == 0)    *out_loss = *acc * (1.0f / 8388608.0f);
}

// ---------------------------------------------------------------------------
extern "C" void kernel_launch(void* const* d_in, const int* in_sizes, int n_in,
                              void* d_out, int out_size, void* d_ws, size_t ws_size,
                              hipStream_t stream) {
    const float* x        = (const float*)d_in[0];
    const float* eW1      = (const float*)d_in[1];
    const float* eb1      = (const float*)d_in[2];
    const float* eW2      = (const float*)d_in[3];
    const float* eb2      = (const float*)d_in[4];
    const float* eW3      = (const float*)d_in[5];
    const float* eb3      = (const float*)d_in[6];
    const float* dW1      = (const float*)d_in[7];
    const float* db1      = (const float*)d_in[8];
    const float* dW2      = (const float*)d_in[9];
    const float* db2      = (const float*)d_in[10];
    const float* dW3      = (const float*)d_in[11];
    const float* db3      = (const float*)d_in[12];
    const float* codebook = (const float*)d_in[13];

    float* out = (float*)d_out;

    // bf16 workspace region
    __bf16* b = (__bf16*)d_ws;
    __bf16* xb  = b;  b += (size_t)4096 * 1024;
    __bf16* h1b = b;  b += (size_t)4096 * 512;
    __bf16* h2b = b;  b += (size_t)4096 * 512;
    __bf16* zb  = b;  b += (size_t)4096 * 2048;
    __bf16* qb  = b;  b += (size_t)4096 * 2048;
    __bf16* we1 = b;  b += (size_t)1024 * 512;
    __bf16* we2 = b;  b += (size_t)512 * 512;
    __bf16* we3 = b;  b += (size_t)512 * 2048;
    __bf16* wd1 = b;  b += (size_t)2048 * 512;
    __bf16* wd2 = b;  b += (size_t)512 * 512;
    __bf16* wd3 = b;  b += (size_t)512 * 1024;
    __bf16* cbb = b;  b += (size_t)1024 * 128;
    // f32 / int region
    float* cn  = (float*)b;
    float* acc = cn + 1024;
    int*   idxbuf = (int*)(acc + 1);

    auto cvt = [&](const float* s, __bf16* d, size_t n) {
        int nv = (int)(n / 8);
        cvt_bf16<<<(nv + 255) / 256, 256, 0, stream>>>(s, d, nv);
    };

    zero1<<<1, 1, 0, stream>>>(acc);
    cnorm_kernel<<<4, 256, 0, stream>>>(codebook, cn);

    // One-time f32 -> bf16 conversions (activations/weights feed the WMMA path)
    cvt(x,        xb,  (size_t)4096 * 1024);
    cvt(eW1,      we1, (size_t)1024 * 512);
    cvt(eW2,      we2, (size_t)512 * 512);
    cvt(eW3,      we3, (size_t)512 * 2048);
    cvt(dW1,      wd1, (size_t)2048 * 512);
    cvt(dW2,      wd2, (size_t)512 * 512);
    cvt(dW3,      wd3, (size_t)512 * 1024);
    cvt(codebook, cbb, (size_t)1024 * 128);

    // Encoder
    gemm_bias_act<true,  __bf16><<<dim3(512 / TILE_N, 4096 / TILE_M), 256, 0, stream>>>(xb,  we1, eb1, h1b, 4096, 512, 1024);
    gemm_bias_act<true,  __bf16><<<dim3(512 / TILE_N, 4096 / TILE_M), 256, 0, stream>>>(h1b, we2, eb2, h2b, 4096, 512, 512);
    gemm_bias_act<false, __bf16><<<dim3(2048 / TILE_N, 4096 / TILE_M), 256, 0, stream>>>(h2b, we3, eb3, zb, 4096, 2048, 512);

    // VQ
    vq_argmin<<<65536 / 128, 256, 0, stream>>>(zb, cbb, cn, idxbuf);
    quant_commit<<<(65536 * 128) / 256, 256, 0, stream>>>(zb, codebook, idxbuf, qb, acc);

    // Decoder (straight-through forward: input = quantized)
    gemm_bias_act<true,  __bf16><<<dim3(512 / TILE_N, 4096 / TILE_M), 256, 0, stream>>>(qb,  wd1, db1, h1b, 4096, 512, 2048);
    gemm_bias_act<true,  __bf16><<<dim3(512 / TILE_N, 4096 / TILE_M), 256, 0, stream>>>(h1b, wd2, db2, h2b, 4096, 512, 512);
    gemm_bias_act<false, float ><<<dim3(1024 / TILE_N, 4096 / TILE_M), 256, 0, stream>>>(h2b, wd3, db3, out, 4096, 1024, 512);

    // Tuple tail: indices (as float) then commit_loss
    finalize<<<65536 / 256, 256, 0, stream>>>(idxbuf, acc,
                                              out + 4096 * 1024,
                                              out + 4096 * 1024 + 65536);
}
